// GPTModel_12077448037099
// MI455X (gfx1250) — matrix operand order repaired and checked by reading, not compiled, not run on hardware
//
#include <hip/hip_runtime.h>
#include <hip/hip_bf16.h>

// ---------------- model constants ----------------
#define DMODEL 768
#define SEQLEN 1024
#define BATCH  4
#define NHEAD  12
#define HDIM   64
#define NLAYER 12
#define FFND   3072
#define VOCAB  50257
#define NTOK   (BATCH * SEQLEN)

typedef __attribute__((ext_vector_type(16))) __bf16 v16bf;
typedef __attribute__((ext_vector_type(8)))  float  v8f;

// Async global->LDS copy path (CDNA5 GLOBAL_LOAD_ASYNC_TO_LDS_B128), guarded
// so we fall back to global_load + ds_store if the toolchain lacks the builtin.
#if defined(__has_builtin)
#  if __has_builtin(__builtin_amdgcn_global_load_async_to_lds_b128)
#    define USE_ASYNC_COPY 1
#  endif
#endif
#ifndef USE_ASYNC_COPY
#  define USE_ASYNC_COPY 0
#endif

#if USE_ASYNC_COPY
// Exact parameter types per hipcc diagnostic:
//   param1: int __attribute__((vector_size(16))) __device__*  (global, AS1)
//   param2: LDS counterpart (AS3)
typedef int v4i_vs __attribute__((vector_size(16)));
typedef __attribute__((address_space(1))) v4i_vs* gv4i_p;
typedef __attribute__((address_space(3))) v4i_vs* sv4i_p;
#endif

// ---------------- WMMA helpers ----------------
__device__ inline v8f wmma_bf16(v16bf a, v16bf b, v8f c) {
    // D(16x16 f32) = A(16x32 bf16) * B(32x16 bf16) + C
    return __builtin_amdgcn_wmma_f32_16x16x32_bf16(
        false, a, false, b, (short)0, c, false, false);
}

// Load a 16-element bf16 fragment as two 8-element (16B) chunks.
__device__ inline v16bf load_frag2(const __bf16* p0, const __bf16* p1) {
    union { uint4 u[2]; v16bf v; } f;
    f.u[0] = *reinterpret_cast<const uint4*>(p0);
    f.u[1] = *reinterpret_cast<const uint4*>(p1);
    return f.v;
}

// ---------------- elementwise kernels ----------------
__global__ __launch_bounds__(256) void k_embed(const int* __restrict__ idx,
                                               const float* __restrict__ tok,
                                               const float* __restrict__ pos,
                                               float* __restrict__ X) {
    int i = blockIdx.x * 256 + threadIdx.x;
    if (i >= NTOK * DMODEL) return;
    int t = i / DMODEL, d = i - t * DMODEL;
    int s = t & (SEQLEN - 1);
    X[i] = tok[(size_t)idx[t] * DMODEL + d] + pos[(size_t)s * DMODEL + d];
}

// W fp32 row-major [K,N]  ->  Wt bf16 row-major [N,K]  (transposed convert)
__global__ __launch_bounds__(256) void k_convT(const float* __restrict__ W,
                                               __bf16* __restrict__ Wt,
                                               int K, int N) {
    size_t i = (size_t)blockIdx.x * 256 + threadIdx.x;
    if (i >= (size_t)K * N) return;
    int n = (int)(i / (size_t)K);
    int k = (int)(i - (size_t)n * K);
    Wt[i] = (__bf16)W[(size_t)k * N + n];
}

// V bf16 [tok, D] -> Vt bf16 [b, h, d, s]
__global__ __launch_bounds__(256) void k_transpV(const __bf16* __restrict__ V,
                                                 __bf16* __restrict__ Vt) {
    int i = blockIdx.x * 256 + threadIdx.x;
    if (i >= NTOK * DMODEL) return;
    int t = i / DMODEL, d = i - t * DMODEL;
    int b = t / SEQLEN, s = t - b * SEQLEN;
    int h = d / HDIM, dd = d - h * HDIM;
    Vt[(((size_t)(b * NHEAD + h)) * HDIM + dd) * SEQLEN + s] = V[i];
}

// LayerNorm over D=768, fp32 in -> bf16 out. One 256-thread block per row.
__global__ __launch_bounds__(256) void k_layernorm(const float* __restrict__ X,
                                                   const float* __restrict__ g,
                                                   const float* __restrict__ bta,
                                                   __bf16* __restrict__ out) {
    __shared__ float s1[256], s2[256];
    const int r = blockIdx.x;
    const float* xr = X + (size_t)r * DMODEL;
    float xv[3], lsum = 0.f, lsq = 0.f;
#pragma unroll
    for (int i = 0; i < 3; ++i) {
        xv[i] = xr[threadIdx.x + i * 256];
        lsum += xv[i]; lsq += xv[i] * xv[i];
    }
    s1[threadIdx.x] = lsum; s2[threadIdx.x] = lsq;
    __syncthreads();
    for (int off = 128; off > 0; off >>= 1) {
        if (threadIdx.x < off) {
            s1[threadIdx.x] += s1[threadIdx.x + off];
            s2[threadIdx.x] += s2[threadIdx.x + off];
        }
        __syncthreads();
    }
    const float mean = s1[0] * (1.f / DMODEL);
    const float var  = s2[0] * (1.f / DMODEL) - mean * mean;
    const float rstd = rsqrtf(var + 1e-5f);
#pragma unroll
    for (int i = 0; i < 3; ++i) {
        int c = threadIdx.x + i * 256;
        out[(size_t)r * DMODEL + c] = (__bf16)((xv[i] - mean) * rstd * g[c] + bta[c]);
    }
}

// ---------------- WMMA GEMM with LDS-staged B ----------------
// C[M,N] = act( A[M,K]bf16 * Bt[N,K]bf16^T + bias ) + resid
// Block = 256 threads = 8 waves. Block tile 256x64: each wave owns 32x64
// (2 A-fragments x 4 B-fragments = 8 WMMAs per 32-wide K step).
// The shared 64x32 B chunk is staged ONCE per block into LDS (async copy on
// CDNA5 when available), killing the 8x redundant B reads of the naive form.
// LDS layout: 64 cols, stride 40 bf16 (80B) -> conflict-free ds_load_b128.
template <int GELU_ACT, int OUT_BF16>
__global__ __launch_bounds__(256) void k_gemm(const __bf16* __restrict__ A,
                                              const __bf16* __restrict__ Bt,
                                              const float* __restrict__ bias,
                                              const float* __restrict__ resid,
                                              float* __restrict__ Cf,
                                              __bf16* __restrict__ Cb,
                                              int M, int N, int K) {
    __shared__ __bf16 ldsB[64 * 40];

    const int lane = threadIdx.x & 31;
    const int wave = threadIdx.x >> 5;
    const int hlf  = lane >> 4;   // lane half (K-interleave select)
    const int lm   = lane & 15;
    const int mbase = blockIdx.y * 256 + wave * 32;
    const int nbase = blockIdx.x * 64;

    // staging assignment: thread copies one 16B chunk per K-step
    const int srow  = threadIdx.x >> 2;          // 0..63 : column in 64-wide tile
    const int spart = threadIdx.x & 3;           // 0..3  : 8-elem chunk of 32 K
    const int sng   = nbase + srow;
    __bf16* sdst = ldsB + srow * 40 + spart * 8;

    v8f zero = {0.f, 0.f, 0.f, 0.f, 0.f, 0.f, 0.f, 0.f};
    v8f acc[2][4];
#pragma unroll
    for (int mt = 0; mt < 2; ++mt)
#pragma unroll
        for (int c = 0; c < 4; ++c) acc[mt][c] = zero;

    const __bf16* arow0 = A + (size_t)(mbase + lm) * K;
    const __bf16* arow1 = A + (size_t)(mbase + 16 + lm) * K;

    for (int kb = 0; kb < K; kb += 32) {
        // ---- stage B chunk (64 cols x 32 K) into LDS ----
        if (sng < N) {
            const __bf16* src = Bt + (size_t)sng * K + kb + spart * 8;
#if USE_ASYNC_COPY
            __builtin_amdgcn_global_load_async_to_lds_b128(
                (gv4i_p)src, (sv4i_p)sdst, 0, 0);
#else
            *reinterpret_cast<uint4*>(sdst) =
                *reinterpret_cast<const uint4*>(src);
#endif
        } else {
            *reinterpret_cast<uint4*>(sdst) = make_uint4(0u, 0u, 0u, 0u);
        }
#if USE_ASYNC_COPY
#  if __has_builtin(__builtin_amdgcn_s_wait_asynccnt)
        __builtin_amdgcn_s_wait_asynccnt(0);
#  else
        asm volatile("s_wait_asynccnt 0x0" ::: "memory");
#  endif
#endif
        __syncthreads();

        // A fragments: rows mbase+lm and mbase+16+lm,
        // K pattern {off..off+7, 16+off..16+off+7}, off = hlf*8
        v16bf af0 = load_frag2(arow0 + kb + hlf * 8, arow0 + kb + 16 + hlf * 8);
        v16bf af1 = load_frag2(arow1 + kb + hlf * 8, arow1 + kb + 16 + hlf * 8);
#pragma unroll
        for (int c = 0; c < 4; ++c) {
            const __bf16* bp = ldsB + (c * 16 + lm) * 40 + hlf * 8;
            v16bf bf = load_frag2(bp, bp + 16);
            acc[0][c] = wmma_bf16(af0, bf, acc[0][c]);
            acc[1][c] = wmma_bf16(af1, bf, acc[1][c]);
        }
        __syncthreads();
    }

#pragma unroll
    for (int mt = 0; mt < 2; ++mt) {
#pragma unroll
        for (int c = 0; c < 4; ++c) {
            const int col = nbase + c * 16 + lm;
            if (col >= N) continue;
            const float bs = bias ? bias[col] : 0.f;
#pragma unroll
            for (int j = 0; j < 8; ++j) {
                const int row = mbase + mt * 16 + hlf * 8 + j;
                float v = acc[mt][c][j] + bs;
                if (GELU_ACT) {
                    float t = 0.7978845608028654f * (v + 0.044715f * v * v * v);
                    v = 0.5f * v * (1.f + tanhf(t));
                }
                if (resid) v += resid[(size_t)row * N + col];
                if (OUT_BF16) Cb[(size_t)row * N + col] = (__bf16)v;
                else          Cf[(size_t)row * N + col] = v;
            }
        }
    }
}

// ---------------- flash attention (wave-level, WMMA) ----------------
// Block = 128 threads = 4 waves; each wave owns one (b, h, 16-row q-tile).
// K/V streamed in 32-wide chunks; online softmax via per-wave LDS scoreboard.
#define MASKVAL (-1.0e30f)
__global__ __launch_bounds__(128) void k_attention(const __bf16* __restrict__ Q,
                                                   const __bf16* __restrict__ Kc,
                                                   const __bf16* __restrict__ Vt,
                                                   __bf16* __restrict__ CTX) {
    __shared__ float sS[4][16 * 33];      // per-wave score tile [16 rows][32 cols]
    __shared__ float sM[4][16], sL[4][16], sF[4][16];

    const int lane = threadIdx.x & 31;
    const int wid  = threadIdx.x >> 5;
    const int hlf  = lane >> 4;
    const int lm   = lane & 15;

    const int id = blockIdx.x * 4 + wid;        // 0..3071
    const int qt = id & 63;
    const int h  = (id >> 6) % NHEAD;
    const int b  = id / (64 * NHEAD);
    const int qbase = qt * 16;
    const int qmax  = qbase + 15;

    // Q fragments (16x64 -> two 16x32 A-operands)
    const __bf16* qp = Q + ((size_t)(b * SEQLEN + qbase + lm)) * DMODEL + h * HDIM;
    v16bf aq0 = load_frag2(qp + hlf * 8,      qp + 16 + hlf * 8);
    v16bf aq1 = load_frag2(qp + 32 + hlf * 8, qp + 48 + hlf * 8);

    v8f zero = {0.f, 0.f, 0.f, 0.f, 0.f, 0.f, 0.f, 0.f};
    v8f O[4]; O[0] = zero; O[1] = zero; O[2] = zero; O[3] = zero;
    float* S = sS[wid];
    if (lane < 16) { sM[wid][lane] = MASKVAL; sL[wid][lane] = 0.f; }

    const __bf16* vhead = Vt + ((size_t)(b * NHEAD + h)) * HDIM * SEQLEN;

    for (int kv0 = 0; kv0 <= qmax; kv0 += 32) {
        // ---- stage 1: raw scores for two 16-kv subtiles into LDS ----
#pragma unroll
        for (int sub = 0; sub < 2; ++sub) {
            const int kvb = kv0 + sub * 16;
            if (kvb <= qmax) {
                const __bf16* kp = Kc + ((size_t)(b * SEQLEN + kvb + lm)) * DMODEL
                                   + h * HDIM + hlf * 8;
                v16bf bk0 = load_frag2(kp,      kp + 16);   // features 0..31
                v16bf bk1 = load_frag2(kp + 32, kp + 48);   // features 32..63
                v8f s = zero;
                s = wmma_bf16(aq0, bk0, s);
                s = wmma_bf16(aq1, bk1, s);
#pragma unroll
                for (int j = 0; j < 8; ++j) {
                    const int row = hlf * 8 + j;
                    const int q  = qbase + row;
                    const int kv = kvb + lm;
                    float v = s[j] * 0.125f;                // 1/sqrt(64)
                    if (kv > q) v = MASKVAL;
                    S[row * 33 + sub * 16 + lm] = v;
                }
            } else {
                for (int idx = lane; idx < 256; idx += 32)
                    S[(idx >> 4) * 33 + 16 + (idx & 15)] = MASKVAL;
            }
        }
        // ---- stage 2: online softmax (lanes 0..15, one row each) ----
        if (lane < 16) {
            const int row = lane;
            float mold = sM[wid][row];
            float cmax = MASKVAL;
#pragma unroll
            for (int t = 0; t < 32; ++t) cmax = fmaxf(cmax, S[row * 33 + t]);
            const float mnew = fmaxf(mold, cmax);
            const float corr = __expf(mold - mnew);
            float sum = 0.f;
#pragma unroll
            for (int t = 0; t < 32; ++t) {
                float p = __expf(S[row * 33 + t] - mnew);
                S[row * 33 + t] = p;
                sum += p;
            }
            sM[wid][row] = mnew;
            sL[wid][row] = sL[wid][row] * corr + sum;
            sF[wid][row] = corr;
        }
        // ---- stage 3: rescale O, P(16x32) x V(32x64) ----
#pragma unroll
        for (int j = 0; j < 8; ++j) {
            const float f = sF[wid][hlf * 8 + j];
            O[0][j] *= f; O[1][j] *= f; O[2][j] *= f; O[3][j] *= f;
        }
        v16bf pf;
#pragma unroll
        for (int i = 0; i < 16; ++i) {
            const int k = (i < 8 ? i : i + 8) + hlf * 8;     // A-operand K pattern
            pf[i] = (__bf16)S[lm * 33 + k];
        }
#pragma unroll
        for (int c = 0; c < 4; ++c) {
            const __bf16* vp = vhead + (size_t)(c * 16 + lm) * SEQLEN + kv0 + hlf * 8;
            v16bf bv = load_frag2(vp, vp + 16);
            O[c] = wmma_bf16(pf, bv, O[c]);
        }
    }

    // ---- finalize: divide by l, write bf16 ctx [tok, D] ----
#pragma unroll
    for (int j = 0; j < 8; ++j) {
        const int row = hlf * 8 + j;
        const float inv = 1.f / sL[wid][row];
        const size_t obase = ((size_t)(b * SEQLEN + qbase + row)) * DMODEL + h * HDIM;
#pragma unroll
        for (int c = 0; c < 4; ++c)
            CTX[obase + c * 16 + lm] = (__bf16)(O[c][j] * inv);
    }
}

// ---------------- host orchestration ----------------
static inline int ceil_div(long long a, long long b) { return (int)((a + b - 1) / b); }

extern "C" void kernel_launch(void* const* d_in, const int* in_sizes, int n_in,
                              void* d_out, int out_size, void* d_ws, size_t ws_size,
                              hipStream_t stream) {
    (void)in_sizes; (void)n_in; (void)out_size; (void)ws_size;
    const int*   in_idx  = (const int*)  d_in[0];
    const float* tok_emb = (const float*)d_in[1];
    const float* pos_emb = (const float*)d_in[2];
    const float* ln1_g   = (const float*)d_in[3];
    const float* ln1_b   = (const float*)d_in[4];
    const float* wq      = (const float*)d_in[5];
    const float* wk      = (const float*)d_in[6];
    const float* wv      = (const float*)d_in[7];
    const float* wo      = (const float*)d_in[8];
    const float* bo      = (const float*)d_in[9];
    const float* ln2_g   = (const float*)d_in[10];
    const float* ln2_b   = (const float*)d_in[11];
    const float* w1      = (const float*)d_in[12];
    const float* b1      = (const float*)d_in[13];
    const float* w2      = (const float*)d_in[14];
    const float* b2      = (const float*)d_in[15];
    const float* lnf_g   = (const float*)d_in[16];
    const float* lnf_b   = (const float*)d_in[17];
    const float* w_out   = (const float*)d_in[18];

    // workspace carve-out (256B aligned)
    char* ws = (char*)d_ws;
    size_t off = 0;
    auto carve = [&](size_t bytes) -> char* {
        char* p = ws + off;
        off += (bytes + 255) & ~(size_t)255;
        return p;
    };
    float*  X    = (float*) carve((size_t)NTOK * DMODEL * 4);
    __bf16* Hb   = (__bf16*)carve((size_t)NTOK * DMODEL * 2);
    __bf16* Qb   = (__bf16*)carve((size_t)NTOK * DMODEL * 2);
    __bf16* Kb   = (__bf16*)carve((size_t)NTOK * DMODEL * 2);
    __bf16* Vb   = (__bf16*)carve((size_t)NTOK * DMODEL * 2);
    __bf16* Vtb  = (__bf16*)carve((size_t)NTOK * DMODEL * 2);
    __bf16* CTXb = (__bf16*)carve((size_t)NTOK * DMODEL * 2);
    __bf16* Fb   = (__bf16*)carve((size_t)NTOK * FFND * 2);
    __bf16* WtQ  = (__bf16*)carve((size_t)DMODEL * DMODEL * 2);
    __bf16* WtK  = (__bf16*)carve((size_t)DMODEL * DMODEL * 2);
    __bf16* WtV  = (__bf16*)carve((size_t)DMODEL * DMODEL * 2);
    __bf16* WtO  = (__bf16*)carve((size_t)DMODEL * DMODEL * 2);
    __bf16* Wt1  = (__bf16*)carve((size_t)FFND * DMODEL * 2);
    __bf16* Wt2  = (__bf16*)carve((size_t)DMODEL * FFND * 2);
    __bf16* WtL  = (__bf16*)carve((size_t)VOCAB * DMODEL * 2);

    const size_t DD = (size_t)DMODEL * DMODEL;
    const size_t DF = (size_t)DMODEL * FFND;

    // embeddings + early (independent) logits-weight conversion
    k_embed<<<ceil_div((long long)NTOK * DMODEL, 256), 256, 0, stream>>>(
        in_idx, tok_emb, pos_emb, X);
    k_convT<<<ceil_div((long long)DMODEL * VOCAB, 256), 256, 0, stream>>>(
        w_out, WtL, DMODEL, VOCAB);

    const dim3 gB(256);
    const dim3 gemmD(ceil_div(DMODEL, 64), NTOK / 256);   // N=768
    const dim3 gemmF(ceil_div(FFND, 64),  NTOK / 256);    // N=3072
    const dim3 gemmV(ceil_div(VOCAB, 64), NTOK / 256);    // N=50257

    for (int l = 0; l < NLAYER; ++l) {
        // transposed bf16 weights for this layer (scratch reused each layer)
        k_convT<<<ceil_div(DD, 256), gB, 0, stream>>>(wq + l * DD, WtQ, DMODEL, DMODEL);
        k_convT<<<ceil_div(DD, 256), gB, 0, stream>>>(wk + l * DD, WtK, DMODEL, DMODEL);
        k_convT<<<ceil_div(DD, 256), gB, 0, stream>>>(wv + l * DD, WtV, DMODEL, DMODEL);
        k_convT<<<ceil_div(DD, 256), gB, 0, stream>>>(wo + l * DD, WtO, DMODEL, DMODEL);
        k_convT<<<ceil_div(DF, 256), gB, 0, stream>>>(w1 + l * DF, Wt1, DMODEL, FFND);
        k_convT<<<ceil_div(DF, 256), gB, 0, stream>>>(w2 + l * DF, Wt2, FFND, DMODEL);

        // LN1 -> bf16
        k_layernorm<<<NTOK, gB, 0, stream>>>(X, ln1_g + l * DMODEL, ln1_b + l * DMODEL, Hb);
        // Q, K, V projections (no bias in reference)
        k_gemm<0, 1><<<gemmD, gB, 0, stream>>>(Hb, WtQ, nullptr, nullptr,
                                               (float*)nullptr, Qb, NTOK, DMODEL, DMODEL);
        k_gemm<0, 1><<<gemmD, gB, 0, stream>>>(Hb, WtK, nullptr, nullptr,
                                               (float*)nullptr, Kb, NTOK, DMODEL, DMODEL);
        k_gemm<0, 1><<<gemmD, gB, 0, stream>>>(Hb, WtV, nullptr, nullptr,
                                               (float*)nullptr, Vb, NTOK, DMODEL, DMODEL);
        // V -> [b,h,d,s] for contiguous B-operand fragments
        k_transpV<<<ceil_div((long long)NTOK * DMODEL, 256), gB, 0, stream>>>(Vb, Vtb);
        // flash attention: B*H*(S/16) wave-tiles, 4 per block
        k_attention<<<(BATCH * NHEAD * (SEQLEN / 16)) / 4, 128, 0, stream>>>(Qb, Kb, Vtb, CTXb);
        // output projection + bo + residual (fp32 X updated in place)
        k_gemm<0, 0><<<gemmD, gB, 0, stream>>>(CTXb, WtO, bo + l * DMODEL, X,
                                               X, (__bf16*)nullptr, NTOK, DMODEL, DMODEL);
        // LN2 -> bf16
        k_layernorm<<<NTOK, gB, 0, stream>>>(X, ln2_g + l * DMODEL, ln2_b + l * DMODEL, Hb);
        // FFN up + GELU -> bf16
        k_gemm<1, 1><<<gemmF, gB, 0, stream>>>(Hb, Wt1, b1 + l * FFND, nullptr,
                                               (float*)nullptr, Fb, NTOK, FFND, DMODEL);
        // FFN down + b2 + residual
        k_gemm<0, 0><<<gemmD, gB, 0, stream>>>(Fb, Wt2, b2 + l * DMODEL, X,
                                               X, (__bf16*)nullptr, NTOK, DMODEL, FFND);
    }

    // final LN + logits GEMM [4096,768] x [768,50257]
    k_layernorm<<<NTOK, gB, 0, stream>>>(X, lnf_g, lnf_b, Hb);
    k_gemm<0, 0><<<gemmV, gB, 0, stream>>>(Hb, WtL, nullptr, nullptr,
                                           (float*)d_out, (__bf16*)nullptr,
                                           NTOK, VOCAB, DMODEL);
}
